// ConvolutionalMem_37452114821969
// MI455X (gfx1250) — matrix operand order repaired and checked by reading, not compiled, not run on hardware
//
#include <hip/hip_runtime.h>
#include <hip/hip_bf16.h>

// ---------------- model dims ----------------
constexpr int Bk   = 32;            // batch
constexpr int CXc  = 3;             // x channels
constexpr int XSc  = 64;            // x size
constexpr int CYc  = 32;            // y channels
constexpr int Kc   = 10;            // kernel
constexpr int Mc   = 55;            // y size
constexpr int Pc   = Mc * Mc;       // 3025 pixels
constexpr int FEATc = CYc * Pc;     // 96800
constexpr int Zc   = 256;
constexpr int STEPSc = 30;
constexpr float DT_TAU_Y = 0.5f;    // 0.1 / 0.2
constexpr float DT_TAU_X = 0.1f;    // 0.1 / 1.0
constexpr float BETA_Y = 3.0f;
constexpr float BETA_Z = 7.0f;

// split-K for the z-GEMM: 3025 k-iters = 55 chunks * 55 iters
constexpr int ZK_SPLIT = 55;
constexpr int ZK_ITERS = 55;

typedef __attribute__((ext_vector_type(16))) _Float16 v16h;
typedef __attribute__((ext_vector_type(8)))  float    v8f;
union U16h { v16h v; _Float16 e[16]; };
union U8f  { v8f  v; float    e[8];  };

// ---------------- W fp32 -> f16 (row-major + transposed) ----------------
__global__ __launch_bounds__(256)
void k_convert_w(const float* __restrict__ W, _Float16* __restrict__ Wh,
                 _Float16* __restrict__ Wt) {
    int idx = blockIdx.x * blockDim.x + threadIdx.x;
    if (idx >= Zc * FEATc) return;
    int zi = idx / FEATc;
    int f  = idx - zi * FEATc;
    _Float16 v = (_Float16)W[idx];
    Wh[idx] = v;                            // [Z, FEAT]
    Wt[(size_t)f * Zc + zi] = v;            // [FEAT, Z]
}

// ---------------- softmax over channel dim ----------------
template <typename OUT>
__global__ __launch_bounds__(256)
void k_softmax_ch(const float* __restrict__ y, OUT* __restrict__ out, float beta) {
    int idx = blockIdx.x * blockDim.x + threadIdx.x;
    if (idx >= Bk * Pc) return;
    int b = idx / Pc;
    int p = idx - b * Pc;
    const float* base = y + (size_t)b * FEATc + p;
    float mx = -1e30f;
    #pragma unroll 4
    for (int c = 0; c < CYc; ++c) mx = fmaxf(mx, base[c * Pc]);
    float s = 0.f;
    #pragma unroll 4
    for (int c = 0; c < CYc; ++c) s += __expf(beta * (base[c * Pc] - mx));
    float inv = 1.f / s;
    OUT* ob = out + (size_t)b * FEATc + p;
    #pragma unroll 4
    for (int c = 0; c < CYc; ++c)
        ob[c * Pc] = (OUT)(__expf(beta * (base[c * Pc] - mx)) * inv);
}

// ---------------- z = W @ fy^T  (split-K WMMA f16, atomic reduce) -------
// Wh: [Z, FEAT] f16;  fy: [B, FEAT] f16;  z: [Z, B] f32 (pre-zeroed)
// block = 64 threads = 2 waves; wave w handles batch-tile w (b0 = 16*w)
__global__ __launch_bounds__(64)
void k_gemm_z(const _Float16* __restrict__ Wh, const _Float16* __restrict__ fy,
              float* __restrict__ z) {
    const int l      = threadIdx.x & 31;
    const int wave   = threadIdx.x >> 5;
    const int z0     = blockIdx.x * 16;          // 16 z-tiles
    const int b0     = wave * 16;                // 2 batch tiles
    const int kchunk = blockIdx.y;               // 55 split-K chunks
    const int row    = l & 15;
    const int hi     = (l >> 4) & 1;
    const int kbA    = hi ? 8 : 0;               // A f16 16x32 lane layout
    const int kbB    = hi ? 16 : 0;              // unused; B is loaded with kbA, same k order as A

    const _Float16* arow = Wh + (size_t)(z0 + row) * FEATc;
    const _Float16* brow = fy + (size_t)(b0 + row) * FEATc;

    v8f acc = {};
    int f0 = kchunk * ZK_ITERS * 32;
    for (int it = 0; it < ZK_ITERS; ++it, f0 += 32) {
        __builtin_prefetch(arow + f0 + 128, 0, 1);
        __builtin_prefetch(brow + f0 + 128, 0, 1);
        U16h a, bb;
        #pragma unroll
        for (int h = 0; h < 8; ++h) {
            a.e[h]     = arow[f0 + kbA + h];          // K = kbA..kbA+7
            a.e[8 + h] = arow[f0 + 16 + kbA + h];     // K = 16+kbA..
        }
        #pragma unroll
        for (int h = 0; h < 8; ++h) {
            bb.e[h]     = brow[f0 + kbA + h];         // K = kbA..kbA+7
            bb.e[8 + h] = brow[f0 + 16 + kbA + h];    // K = 16+kbA..
        }
        acc = __builtin_amdgcn_wmma_f32_16x16x32_f16(
            false, a.v, false, bb.v, (short)0, acc, false, false);
    }
    U8f d; d.v = acc;
    const int n = l & 15;
    const int mbase = hi ? 8 : 0;
    #pragma unroll
    for (int r = 0; r < 8; ++r)
        atomicAdd(&z[(size_t)(z0 + mbase + r) * Bk + b0 + n], d.e[r]);
}

// ---------------- fz = softmax_z(7*z), per batch row ----------------
// z: [Z, B] f32  ->  fz: [B, Z] f16.  One wave per batch row.
__global__ __launch_bounds__(32)
void k_softmax_z(const float* __restrict__ z, _Float16* __restrict__ fz) {
    int b = blockIdx.x;
    int t = threadIdx.x;
    float vals[8];
    float mx = -1e30f;
    #pragma unroll
    for (int j = 0; j < 8; ++j) {
        float v = BETA_Z * z[(size_t)(t + 32 * j) * Bk + b];
        vals[j] = v;
        mx = fmaxf(mx, v);
    }
    #pragma unroll
    for (int off = 16; off > 0; off >>= 1) mx = fmaxf(mx, __shfl_xor(mx, off, 32));
    float s = 0.f;
    #pragma unroll
    for (int j = 0; j < 8; ++j) { vals[j] = __expf(vals[j] - mx); s += vals[j]; }
    #pragma unroll
    for (int off = 16; off > 0; off >>= 1) s += __shfl_xor(s, off, 32);
    float inv = 1.f / s;
    #pragma unroll
    for (int j = 0; j < 8; ++j)
        fz[(size_t)b * Zc + t + 32 * j] = (_Float16)(vals[j] * inv);
}

// ---------------- conv_x = conv2d(x, Kw) VALID ----------------
__global__ __launch_bounds__(256)
void k_conv(const float* __restrict__ x, const float* __restrict__ Kw,
            float* __restrict__ cx) {
    int idx = blockIdx.x * blockDim.x + threadIdx.x;
    if (idx >= Bk * CYc * Pc) return;
    int b  = idx / (CYc * Pc);
    int r  = idx - b * CYc * Pc;
    int co = r / Pc;
    int p  = r - co * Pc;
    int i  = p / Mc;
    int j  = p - i * Mc;
    const float* xb = x  + (size_t)b  * CXc * XSc * XSc;
    const float* kb = Kw + (size_t)co * CXc * Kc  * Kc;
    float s = 0.f;
    for (int ci = 0; ci < CXc; ++ci) {
        const float* xc = xb + (size_t)ci * XSc * XSc + i * XSc + j;
        const float* kc = kb + ci * Kc * Kc;
        for (int ki = 0; ki < Kc; ++ki) {
            #pragma unroll
            for (int kj = 0; kj < Kc; ++kj)
                s = fmaf(xc[ki * XSc + kj], kc[ki * Kc + kj], s);
        }
    }
    cx[idx] = s;
}

// ---------------- top = fz @ W, fused y-EMA update (WMMA f16) ----------
// fz: [B, Z] f16; Wt: [FEAT, Z] f16; cx: [B, FEAT] f32; y: [B, FEAT] f32
// block = 64 = 2 waves; wave w handles f-tile (2*blockIdx.x + w)
__global__ __launch_bounds__(64)
void k_gemm_top(const _Float16* __restrict__ fz, const _Float16* __restrict__ Wt,
                const float* __restrict__ cx, float* __restrict__ y) {
    const int l    = threadIdx.x & 31;
    const int wave = threadIdx.x >> 5;
    const int f0   = (blockIdx.x * 2 + wave) * 16;   // 6050 f-tiles
    const int b0   = blockIdx.y * 16;                // 2 batch tiles
    const int row  = l & 15;
    const int hi   = (l >> 4) & 1;
    const int kbA  = hi ? 8 : 0;
    const int kbB  = hi ? 16 : 0;

    const _Float16* arow = fz + (size_t)(b0 + row) * Zc;   // A: [16 x 32] of fz
    const _Float16* brow = Wt + (size_t)(f0 + row) * Zc;   // B: [32 x 16] of W^T

    v8f acc = {};
    #pragma unroll
    for (int kz = 0; kz < Zc / 32; ++kz) {
        const int zz = kz * 32;
        U16h a, bb;
        #pragma unroll
        for (int h = 0; h < 8; ++h) {
            a.e[h]     = arow[zz + kbA + h];
            a.e[8 + h] = arow[zz + 16 + kbA + h];
        }
        #pragma unroll
        for (int h = 0; h < 8; ++h) {
            bb.e[h]     = brow[zz + kbA + h];
            bb.e[8 + h] = brow[zz + 16 + kbA + h];
        }
        acc = __builtin_amdgcn_wmma_f32_16x16x32_f16(
            false, a.v, false, bb.v, (short)0, acc, false, false);
    }
    U8f d; d.v = acc;
    const int n = l & 15;
    const int mbase = hi ? 8 : 0;
    #pragma unroll
    for (int r = 0; r < 8; ++r) {
        size_t idx = (size_t)(b0 + mbase + r) * FEATc + f0 + n;
        float yinst = cx[idx] + d.e[r];
        float yv = y[idx];
        y[idx] = yv + DT_TAU_Y * (yinst - yv);
    }
}

// ---------------- x += 0.1*(-x + convT(sy)), in place ----------------
__global__ __launch_bounds__(256)
void k_convT(const float* __restrict__ sy, const float* __restrict__ Kw,
             float* __restrict__ x) {
    int idx = blockIdx.x * blockDim.x + threadIdx.x;
    if (idx >= Bk * CXc * XSc * XSc) return;
    int b  = idx / (CXc * XSc * XSc);
    int r  = idx - b * CXc * XSc * XSc;
    int cc = r / (XSc * XSc);
    int p2 = r - cc * XSc * XSc;
    int p  = p2 / XSc;
    int q  = p2 - p * XSc;
    int ilo = max(p - (Kc - 1), 0), ihi = min(p, Mc - 1);
    int jlo = max(q - (Kc - 1), 0), jhi = min(q, Mc - 1);
    float s = 0.f;
    for (int cy = 0; cy < CYc; ++cy) {
        const float* sb = sy + (size_t)b * FEATc + (size_t)cy * Pc;
        const float* kb = Kw + ((size_t)cy * CXc + cc) * Kc * Kc;
        for (int i = ilo; i <= ihi; ++i) {
            int ki = p - i;
            for (int j = jlo; j <= jhi; ++j)
                s = fmaf(sb[i * Mc + j], kb[ki * Kc + (q - j)], s);
        }
    }
    float xv = x[idx];
    x[idx] = xv + DT_TAU_X * (s - xv);
}

// ---------------- orchestration ----------------
extern "C" void kernel_launch(void* const* d_in, const int* in_sizes, int n_in,
                              void* d_out, int out_size, void* d_ws, size_t ws_size,
                              hipStream_t stream) {
    (void)in_sizes; (void)n_in; (void)out_size; (void)ws_size;
    const float* x_in = (const float*)d_in[0];   // [B, CX, 64, 64]
    const float* Kw   = (const float*)d_in[1];   // [CY, CX, 10, 10]
    const float* W    = (const float*)d_in[2];   // [Z, FEAT]

    auto al = [](size_t v) { return (v + 255) & ~(size_t)255; };
    char* base = (char*)d_ws;
    size_t o = 0;
    const size_t NW = (size_t)Zc * FEATc;
    _Float16* Wh  = (_Float16*)(base + o); o = al(o + NW * 2);
    _Float16* Wt  = (_Float16*)(base + o); o = al(o + NW * 2);
    _Float16* fy  = (_Float16*)(base + o); o = al(o + (size_t)Bk * FEATc * 2);
    _Float16* fz  = (_Float16*)(base + o); o = al(o + (size_t)Bk * Zc * 2);
    float*    zb  = (float*)(base + o);    o = al(o + (size_t)Zc * Bk * 4);
    float*    cxb = (float*)(base + o);    o = al(o + (size_t)Bk * FEATc * 4);
    float*    yb  = (float*)(base + o);    o = al(o + (size_t)Bk * FEATc * 4);
    float*    syb = (float*)(base + o);    o = al(o + (size_t)Bk * FEATc * 4);
    float*    xb  = (float*)(base + o);    o = al(o + (size_t)Bk * CXc * XSc * XSc * 4);

    const int NX = Bk * CXc * XSc * XSc;     // 393216
    // seed state (deterministic each call)
    hipMemcpyAsync(xb, x_in, (size_t)NX * 4, hipMemcpyDeviceToDevice, stream);
    hipMemsetAsync(yb, 0, (size_t)Bk * FEATc * 4, stream);
    k_convert_w<<<(int)((NW + 255) / 256), 256, 0, stream>>>(W, Wh, Wt);

    const int grdBP  = (Bk * Pc + 255) / 256;           // softmax_ch sites
    const int grdCV  = (Bk * CYc * Pc + 255) / 256;     // conv outputs
    const int grdCT  = (NX + 255) / 256;                // convT outputs

    for (int step = 0; step < STEPSc; ++step) {
        hipMemsetAsync(zb, 0, (size_t)Zc * Bk * 4, stream);
        k_softmax_ch<_Float16><<<grdBP, 256, 0, stream>>>(yb, fy, BETA_Y);
        k_gemm_z<<<dim3(Zc / 16, ZK_SPLIT), 64, 0, stream>>>(Wh, fy, zb);
        k_softmax_z<<<Bk, 32, 0, stream>>>(zb, fz);
        k_conv<<<grdCV, 256, 0, stream>>>(xb, Kw, cxb);
        k_gemm_top<<<dim3(FEATc / 32, Bk / 16), 64, 0, stream>>>(fz, Wt, cxb, yb);
        k_softmax_ch<float><<<grdBP, 256, 0, stream>>>(yb, syb, 1.0f);
        k_convT<<<grdCT, 256, 0, stream>>>(syb, Kw, xb);
    }
    hipMemcpyAsync(d_out, xb, (size_t)NX * 4, hipMemcpyDeviceToDevice, stream);
}
